// CausalSelfAttention_90950227460127
// MI455X (gfx1250) — compile-verified, matrix operand order checked by reading, not run on hardware
//
#include <hip/hip_runtime.h>
#include <hip/hip_bf16.h>

// ---------------------------------------------------------------------------
// CDNA5 (gfx1250) causal GQA self-attention: bf16 WMMA + TDM-staged GEMM.
// ---------------------------------------------------------------------------

typedef __bf16 bf16;
typedef __attribute__((ext_vector_type(16))) __bf16 v16bf;
typedef __attribute__((ext_vector_type(8)))  __bf16 v8bf;
typedef __attribute__((ext_vector_type(8)))  float  v8f;
typedef __attribute__((ext_vector_type(4)))  unsigned int v4u;
typedef __attribute__((ext_vector_type(8)))  int v8i;
typedef __attribute__((ext_vector_type(4)))  int v4i;

#define B_   2
#define T_   2048
#define C_   2048
#define NH_  16
#define NKV_ 4
#define HD_  128
#define MTOK (B_ * T_)

#if defined(__HIP_DEVICE_COMPILE__) && __has_builtin(__builtin_amdgcn_tensor_load_to_lds)
#define HAVE_TDM 1
#else
#define HAVE_TDM 0
#endif

// Load a 16-half fragment: elements 0..7 from base0, 8..15 from base1.
// Matches ISA 16-bit A/B layout: per lane, K = kb..kb+7 then kb+16..kb+23.
__device__ __forceinline__ v16bf load_frag(const bf16* base0, const bf16* base1) {
    v8bf lo = *(const v8bf*)base0;
    v8bf hi = *(const v8bf*)base1;
    v16bf r;
#pragma unroll
    for (int i = 0; i < 8; ++i) { r[i] = lo[i]; r[8 + i] = hi[i]; }
    return r;
}

__device__ __forceinline__ v8f wmma_bf16(v16bf a, v16bf b, v8f c) {
    return __builtin_amdgcn_wmma_f32_16x16x32_bf16(false, a, false, b, (short)0, c,
                                                   false, false);
}

#if HAVE_TDM
// Issue a TDM 2-D tile load: tile_h x tile_w bf16 elements from a row-major
// [tensor_h x tensor_w] matrix (row stride = stride_elems) into LDS at lds_off,
// packed row-major [tile_h][tile_w]. Tracked by TENSORcnt.
__device__ __forceinline__ void tdm_load_2d(unsigned lds_off, const bf16* gptr,
                                            unsigned tensor_w, unsigned tensor_h,
                                            unsigned tile_w, unsigned tile_h,
                                            unsigned stride_elems) {
    unsigned long long ga = (unsigned long long)gptr;
    v4u g0;
    g0[0] = 1u;                                            // count=1, user D#
    g0[1] = lds_off;                                       // LDS byte address
    g0[2] = (unsigned)(ga & 0xffffffffu);                  // global addr lo
    g0[3] = (unsigned)((ga >> 32) & 0x1ffffffu) | (2u << 30); // addr hi | type=2
    v8i g1;
    g1[0] = (int)(1u << 16);                               // data_size=1 (2 B)
    g1[1] = (int)((tensor_w & 0xffffu) << 16);             // tensor_dim0 lo16
    g1[2] = (int)((tensor_w >> 16) | ((tensor_h & 0xffffu) << 16));
    g1[3] = (int)((tensor_h >> 16) | (tile_w << 16));      // tile_dim0
    g1[4] = (int)tile_h;                                   // tile_dim1, tile_dim2=0
    g1[5] = (int)stride_elems;                             // tensor_dim0_stride lo32
    g1[6] = 0;
    g1[7] = 0;
    v4i z4 = {0, 0, 0, 0};
#if __clang_major__ >= 23
    v8i z8 = {0, 0, 0, 0, 0, 0, 0, 0};
    __builtin_amdgcn_tensor_load_to_lds(g0, g1, z4, z4, z8, 0);
#else
    __builtin_amdgcn_tensor_load_to_lds(g0, g1, z4, z4, 0);
#endif
}
#endif

// ---------------------------------------------------------------------------
// 1) f32 -> bf16 convert (same layout)
// ---------------------------------------------------------------------------
__global__ void to_bf16_kernel(const float* __restrict__ in, bf16* __restrict__ out,
                               long n) {
    long i = (long)blockIdx.x * blockDim.x + threadIdx.x;
    if (i < n) out[i] = (bf16)in[i];
}

// ---------------------------------------------------------------------------
// 2) W [K,N] f32 -> Wt [N,K] bf16   (so WMMA B fragments are row loads)
// ---------------------------------------------------------------------------
__global__ void transpose_bf16_kernel(const float* __restrict__ in,
                                      bf16* __restrict__ out, int K, int N) {
    long i = (long)blockIdx.x * blockDim.x + threadIdx.x;
    long n = (long)K * N;
    if (i < n) {
        int k = (int)(i / N), c = (int)(i % N);
        out[(size_t)c * K + k] = (bf16)in[i];
    }
}

// ---------------------------------------------------------------------------
// 3) GEMM: Out[M,N] f32 = A[M,K] bf16 @ Bt[N,K]^T bf16.
//    Block = 4 waves, tile 128(M) x 64(N). A/B K-slices (Kc=32) staged in LDS
//    by the Tensor Data Mover, double-buffered on TENSORcnt. Each wave owns a
//    32x64 register tile: 8 accumulators, 8 WMMAs per K-step.
// ---------------------------------------------------------------------------
__global__ __launch_bounds__(128)
void gemm_bf16_kernel(const bf16* __restrict__ A, const bf16* __restrict__ Bt,
                      float* __restrict__ Out, int M, int N, int K) {
    __shared__ bf16 sA[2][128 * 32];   // [buf][row][k]   8 KB each
    __shared__ bf16 sB[2][64 * 32];    // [buf][col][k]   4 KB each

    const int tid  = threadIdx.x;
    const int wave = tid >> 5;
    const int lane = tid & 31;
    const int ln = lane & 15;
    const int hf = lane >> 4;
    const int kb = hf * 8;

    const int m0 = blockIdx.x * 128;
    const int n0 = blockIdx.y * 64;
    const int nk = K / 32;

    v8f acc0[4], acc1[4];
#pragma unroll
    for (int j = 0; j < 4; ++j)
#pragma unroll
        for (int r = 0; r < 8; ++r) { acc0[j][r] = 0.0f; acc1[j][r] = 0.0f; }

#if HAVE_TDM
    const unsigned ldsA0 = (unsigned)(size_t)&sA[0][0];
    const unsigned ldsA1 = (unsigned)(size_t)&sA[1][0];
    const unsigned ldsB0 = (unsigned)(size_t)&sB[0][0];
    const unsigned ldsB1 = (unsigned)(size_t)&sB[1][0];
    if (wave == 0) {
        tdm_load_2d(ldsA0, A + (size_t)m0 * K, (unsigned)K, (unsigned)M, 32, 128,
                    (unsigned)K);
        tdm_load_2d(ldsB0, Bt + (size_t)n0 * K, (unsigned)K, (unsigned)N, 32, 64,
                    (unsigned)K);
    }
#endif

    int buf = 0;
    for (int ks = 0; ks < nk; ++ks) {
        const int k0 = ks * 32;
#if HAVE_TDM
        if (wave == 0) {
            if (ks + 1 < nk) {
                const int kn = k0 + 32;
                tdm_load_2d(buf ? ldsA0 : ldsA1, A + (size_t)m0 * K + kn,
                            (unsigned)K, (unsigned)M, 32, 128, (unsigned)K);
                tdm_load_2d(buf ? ldsB0 : ldsB1, Bt + (size_t)n0 * K + kn,
                            (unsigned)K, (unsigned)N, 32, 64, (unsigned)K);
                __builtin_amdgcn_s_wait_tensorcnt((short)2);   // stage ks done
            } else {
                __builtin_amdgcn_s_wait_tensorcnt((short)0);
            }
        }
#else
        // Fallback: cooperative global->LDS copy (no TDM builtin available).
        {
            const bf16* ga = A + (size_t)m0 * K + k0;
            for (int i = tid; i < 128 * 4; i += 128) {
                int row = i >> 2, seg = i & 3;
                *(v8bf*)&sA[buf][row * 32 + seg * 8] =
                    *(const v8bf*)(ga + (size_t)row * K + seg * 8);
            }
            const bf16* gb = Bt + (size_t)n0 * K + k0;
            for (int i = tid; i < 64 * 4; i += 128) {
                int row = i >> 2, seg = i & 3;
                *(v8bf*)&sB[buf][row * 32 + seg * 8] =
                    *(const v8bf*)(gb + (size_t)row * K + seg * 8);
            }
        }
#endif
        __syncthreads();

        const bf16* a0 = &sA[buf][(wave * 32 + ln) * 32 + kb];
        const bf16* a1 = &sA[buf][(wave * 32 + 16 + ln) * 32 + kb];
        v16bf af0 = load_frag(a0, a0 + 16);
        v16bf af1 = load_frag(a1, a1 + 16);
#pragma unroll
        for (int j = 0; j < 4; ++j) {
            const bf16* bp = &sB[buf][(j * 16 + ln) * 32 + kb];
            v16bf bfrag = load_frag(bp, bp + 16);
            acc0[j] = wmma_bf16(af0, bfrag, acc0[j]);
            acc1[j] = wmma_bf16(af1, bfrag, acc1[j]);
        }
        __syncthreads();
        buf ^= 1;
    }

    // C/D layout: VGPR r, lane L -> row = r + 8*(L>>4), col = L&15
    const int wm = m0 + wave * 32;
#pragma unroll
    for (int j = 0; j < 4; ++j)
#pragma unroll
        for (int r = 0; r < 8; ++r) {
            Out[(size_t)(wm + r + 8 * hf) * N + n0 + j * 16 + ln]      = acc0[j][r];
            Out[(size_t)(wm + 16 + r + 8 * hf) * N + n0 + j * 16 + ln] = acc1[j][r];
        }
}

// ---------------------------------------------------------------------------
// 4) RoPE + RMSNorm: raw f32 [B,T,nh*HD] -> out bf16 [B,nh,T,HD]
// ---------------------------------------------------------------------------
__global__ __launch_bounds__(32)
void rope_rms_kernel(const float* __restrict__ raw, const float* __restrict__ cosv,
                     const float* __restrict__ sinv, bf16* __restrict__ out, int nh) {
    const int t = blockIdx.x, h = blockIdx.y, b = blockIdx.z, lane = threadIdx.x;
    const float* xr = raw + ((size_t)(b * T_ + t) * nh + h) * HD_;

    float o1[2], o2[2], ss = 0.0f;
#pragma unroll
    for (int i = 0; i < 2; ++i) {
        int p = lane * 2 + i;                      // pair index 0..63
        float x1 = xr[p], x2 = xr[p + 64];
        float c = cosv[t * 64 + p], s = sinv[t * 64 + p];
        o1[i] = x1 * c + x2 * s;
        o2[i] = -x1 * s + x2 * c;
        ss += o1[i] * o1[i] + o2[i] * o2[i];
    }
#pragma unroll
    for (int m = 16; m >= 1; m >>= 1) ss += __shfl_xor(ss, m, 32);
    float rn = rsqrtf(ss * (1.0f / 128.0f) + 1.1920929e-7f);

    bf16* orow = out + ((size_t)(b * nh + h) * T_ + t) * HD_;
#pragma unroll
    for (int i = 0; i < 2; ++i) {
        int p = lane * 2 + i;
        orow[p]      = (bf16)(o1[i] * rn);
        orow[p + 64] = (bf16)(o2[i] * rn);
    }
}

// ---------------------------------------------------------------------------
// 5) V transpose: v_raw f32 [B,T,NKV*HD] -> vt bf16 [B,NKV,HD,T]
// ---------------------------------------------------------------------------
__global__ void vt_kernel(const float* __restrict__ vraw, bf16* __restrict__ vt,
                          long n) {
    long i = (long)blockIdx.x * blockDim.x + threadIdx.x;
    if (i < n) {
        int d = (int)(i % HD_);
        long tmp = i / HD_;
        int h = (int)(tmp % NKV_); tmp /= NKV_;
        int t = (int)(tmp % T_);
        int b = (int)(tmp / T_);
        vt[(((size_t)b * NKV_ + h) * HD_ + d) * T_ + t] = (bf16)vraw[i];
    }
}

// ---------------------------------------------------------------------------
// 6) Flash attention: one wave per (b, h, 16-row q tile).
// ---------------------------------------------------------------------------
__global__ __launch_bounds__(32)
void attn_kernel(const bf16* __restrict__ q, const bf16* __restrict__ k,
                 const bf16* __restrict__ vt, bf16* __restrict__ att) {
    __shared__ bf16 Plds[16 * 32];

    const int lane = threadIdx.x;
    const int t0 = blockIdx.x * 16;
    const int h  = blockIdx.y;
    const int b  = blockIdx.z;
    const int hkv = h / (NH_ / NKV_);
    const int ln = lane & 15;
    const int hf = lane >> 4;
    const int kb = hf * 8;
    const float scale = 0.08838834764831845f;   // 1/sqrt(128)

    const bf16* qbase = q + ((size_t)(b * NH_ + h) * T_ + t0) * HD_;
    v16bf qf[4];
#pragma unroll
    for (int c = 0; c < 4; ++c) {
        const bf16* p = qbase + (size_t)ln * HD_ + c * 32 + kb;
        qf[c] = load_frag(p, p + 16);
    }

    const bf16* kbase = k + (size_t)(b * NKV_ + hkv) * T_ * HD_;
    const bf16* vbase = vt + (size_t)(b * NKV_ + hkv) * HD_ * T_;

    v8f y[8];
    float m[8], l[8];
#pragma unroll
    for (int j = 0; j < 8; ++j) {
#pragma unroll
        for (int r = 0; r < 8; ++r) y[j][r] = 0.0f;
        m[j] = -3.4e38f; l[j] = 0.0f;
    }

    for (int k0 = 0; k0 <= t0 + 15; k0 += 32) {
        v8f s1, s2;
#pragma unroll
        for (int r = 0; r < 8; ++r) { s1[r] = 0.0f; s2[r] = 0.0f; }
#pragma unroll
        for (int c = 0; c < 4; ++c) {
            const bf16* kp1 = kbase + (size_t)(k0 + ln) * HD_ + c * 32 + kb;
            v16bf b1 = load_frag(kp1, kp1 + 16);
            s1 = wmma_bf16(qf[c], b1, s1);
            const bf16* kp2 = kp1 + (size_t)16 * HD_;
            v16bf b2 = load_frag(kp2, kp2 + 16);
            s2 = wmma_bf16(qf[c], b2, s2);
        }

        float f[8];
#pragma unroll
        for (int r = 0; r < 8; ++r) {
            int qrow = t0 + r + 8 * hf;
            float a1 = s1[r] * scale; if (k0 + ln > qrow)      a1 = -3.4e38f;
            float a2 = s2[r] * scale; if (k0 + 16 + ln > qrow) a2 = -3.4e38f;
            float mx = fmaxf(a1, a2);
#pragma unroll
            for (int msk = 8; msk >= 1; msk >>= 1)
                mx = fmaxf(mx, __shfl_xor(mx, msk, 32));
            float mn = fmaxf(m[r], mx);
            float alpha = __expf(m[r] - mn);
            a1 = __expf(a1 - mn);
            a2 = __expf(a2 - mn);
            float rs = a1 + a2;
#pragma unroll
            for (int msk = 8; msk >= 1; msk >>= 1) rs += __shfl_xor(rs, msk, 32);
            l[r] = l[r] * alpha + rs;
            m[r] = mn;
            f[r] = alpha;
            Plds[(r + 8 * hf) * 32 + ln]      = (bf16)a1;
            Plds[(r + 8 * hf) * 32 + 16 + ln] = (bf16)a2;
        }
        __syncthreads();

        const bf16* pp = &Plds[ln * 32 + kb];
        v16bf pf = load_frag(pp, pp + 16);
#pragma unroll
        for (int j = 0; j < 8; ++j) {
#pragma unroll
            for (int r = 0; r < 8; ++r) y[j][r] *= f[r];
            const bf16* vp = vbase + (size_t)(j * 16 + ln) * T_ + k0 + kb;
            v16bf vf = load_frag(vp, vp + 16);
            y[j] = wmma_bf16(pf, vf, y[j]);
        }
        __syncthreads();
    }

    bf16* arow = att + (size_t)(b * T_ + t0) * C_ + h * HD_;
#pragma unroll
    for (int r = 0; r < 8; ++r) {
        float inv = 1.0f / l[r];
        int row = r + 8 * hf;
#pragma unroll
        for (int j = 0; j < 8; ++j)
            arow[(size_t)row * C_ + j * 16 + ln] = (bf16)(y[j][r] * inv);
    }
}

// ---------------------------------------------------------------------------
// Host launcher
// ---------------------------------------------------------------------------
extern "C" void kernel_launch(void* const* d_in, const int* in_sizes, int n_in,
                              void* d_out, int out_size, void* d_ws, size_t ws_size,
                              hipStream_t stream) {
    (void)in_sizes; (void)n_in; (void)out_size; (void)ws_size;

    const float* x   = (const float*)d_in[0];
    const float* cs  = (const float*)d_in[1];
    const float* sn  = (const float*)d_in[2];
    const float* wq  = (const float*)d_in[3];
    const float* wk  = (const float*)d_in[4];
    const float* wv  = (const float*)d_in[5];
    const float* wo  = (const float*)d_in[6];
    float* out       = (float*)d_out;

    char* ws = (char*)d_ws;
    size_t off = 0;
    auto alloc = [&](size_t bytes) { char* p = ws + off; off += (bytes + 255) & ~(size_t)255; return p; };

    bf16*  xb    = (bf16*)alloc((size_t)MTOK * C_ * 2);
    bf16*  wqT   = (bf16*)alloc((size_t)C_ * C_ * 2);
    bf16*  wkT   = (bf16*)alloc((size_t)(NKV_ * HD_) * C_ * 2);
    bf16*  wvT   = (bf16*)alloc((size_t)(NKV_ * HD_) * C_ * 2);
    bf16*  woT   = (bf16*)alloc((size_t)C_ * C_ * 2);
    float* qraw  = (float*)alloc((size_t)MTOK * C_ * 4);
    float* kraw  = (float*)alloc((size_t)MTOK * NKV_ * HD_ * 4);
    float* vraw  = (float*)alloc((size_t)MTOK * NKV_ * HD_ * 4);
    bf16*  qrot  = (bf16*)alloc((size_t)B_ * NH_ * T_ * HD_ * 2);
    bf16*  krot  = (bf16*)alloc((size_t)B_ * NKV_ * T_ * HD_ * 2);
    bf16*  vtp   = (bf16*)alloc((size_t)B_ * NKV_ * HD_ * T_ * 2);
    bf16*  att   = (bf16*)qraw;   // qraw is dead before attn writes att

    const int TB = 256;
    long nx = (long)MTOK * C_;

    to_bf16_kernel<<<(nx + TB - 1) / TB, TB, 0, stream>>>(x, xb, nx);

    transpose_bf16_kernel<<<((long)C_ * C_ + TB - 1) / TB, TB, 0, stream>>>(wq, wqT, C_, C_);
    transpose_bf16_kernel<<<((long)C_ * 512 + TB - 1) / TB, TB, 0, stream>>>(wk, wkT, C_, 512);
    transpose_bf16_kernel<<<((long)C_ * 512 + TB - 1) / TB, TB, 0, stream>>>(wv, wvT, C_, 512);
    transpose_bf16_kernel<<<((long)C_ * C_ + TB - 1) / TB, TB, 0, stream>>>(wo, woT, C_, C_);

    gemm_bf16_kernel<<<dim3(MTOK / 128, C_ / 64), 128, 0, stream>>>(xb, wqT, qraw, MTOK, C_, C_);
    gemm_bf16_kernel<<<dim3(MTOK / 128, 512 / 64), 128, 0, stream>>>(xb, wkT, kraw, MTOK, 512, C_);
    gemm_bf16_kernel<<<dim3(MTOK / 128, 512 / 64), 128, 0, stream>>>(xb, wvT, vraw, MTOK, 512, C_);

    rope_rms_kernel<<<dim3(T_, NH_, B_), 32, 0, stream>>>(qraw, cs, sn, qrot, NH_);
    rope_rms_kernel<<<dim3(T_, NKV_, B_), 32, 0, stream>>>(kraw, cs, sn, krot, NKV_);

    long nv = (long)MTOK * NKV_ * HD_;
    vt_kernel<<<(nv + TB - 1) / TB, TB, 0, stream>>>(vraw, vtp, nv);

    attn_kernel<<<dim3(T_ / 16, NH_, B_), 32, 0, stream>>>(qrot, krot, vtp, att);

    gemm_bf16_kernel<<<dim3(MTOK / 128, C_ / 64), 128, 0, stream>>>(att, woT, out, MTOK, C_, C_);
}
